// DGCNNEncoder_40312563040760
// MI455X (gfx1250) — matrix operand order repaired and checked by reading, not compiled, not run on hardware
//
#include <hip/hip_runtime.h>

// ---------- types ----------
typedef __attribute__((ext_vector_type(16))) __bf16 v16bf;
typedef __attribute__((ext_vector_type(8)))  float  v8f;
typedef unsigned short u16;
typedef unsigned int   u32;

#define KNN 20
#define SLOPE 0.2f

// ---------- bf16 helpers ----------
__device__ __forceinline__ u16 f2bf(float f) {
    u32 u = __float_as_uint(f);
    u32 r = (u + 0x7fffu + ((u >> 16) & 1u)) >> 16;
    return (u16)r;
}
__device__ __forceinline__ float bf2f(u16 h) {
    return __uint_as_float(((u32)h) << 16);
}

__device__ __forceinline__ v8f wmma_bf16(const u32 a[8], const u32 b[8], v8f c) {
    union { u32 u[8]; v16bf v; } ua, ub;
#pragma unroll
    for (int i = 0; i < 8; ++i) { ua.u[i] = a[i]; ub.u[i] = b[i]; }
    return __builtin_amdgcn_wmma_f32_16x16x32_bf16(false, ua.v, false, ub.v,
                                                   (short)0, c, false, false);
}

// ---------- conversion kernels ----------
__global__ __launch_bounds__(256) void f32_to_bf16_kernel(const float* __restrict__ in,
                                                          u16* __restrict__ out, int n) {
    int i = blockIdx.x * 256 + threadIdx.x;
    if (i < n) out[i] = f2bf(in[i]);
}

// x (B,3,N) f32 -> xb (B,N,3) bf16
__global__ __launch_bounds__(256) void x_transpose_kernel(const float* __restrict__ x,
                                                          u16* __restrict__ xb,
                                                          int B, int N) {
    int i = blockIdx.x * 256 + threadIdx.x;
    if (i >= B * N) return;
    int b = i / N, n = i % N;
#pragma unroll
    for (int c = 0; c < 3; ++c)
        xb[(size_t)i * 3 + c] = f2bf(x[((size_t)b * 3 + c) * N + n]);
}

// ---------- kNN: WMMA pairwise inner products + guarded streaming top-20 ----
// fb: (B,N,C) bf16. Rank by (2*inner - |col|^2); the -|row|^2 term is
// row-constant and cannot change per-row ordering, so it is dropped.
// Block = 128 threads (4 waves). Each block: 16 rows x N cols in 64-col tiles.
// Wave w computes the 16x16 pd sub-tile for col group w via WMMA (K = C),
// results staged in LDS; then 16 rows x 8 slots of threads run guarded
// register top-20 insertion; per-row 8-way merge at the end.
template<int C>
__global__ __launch_bounds__(128) void knn_kernel(const u16* __restrict__ fb,
                                                  int* __restrict__ idxout, int B, int N) {
    constexpr int KT  = (C + 31) / 32;
    constexpr int CP2 = C + 2 + (C & 1);            // padded col stride (ushorts)
    constexpr int ROWF = (16 * C + 1) / 2;          // rowsU size in floats
    constexpr int COLF = (64 * CP2 + 1) / 2;        // colsU size in floats
    constexpr int P1 = ROWF + COLF + 64 + 16 * 66;  // + colNorm + pdTile
    constexpr int P2 = 16 * 8 * KNN * 2;            // merge lists (val+idx)
    constexpr int SW = (P1 > P2) ? P1 : P2;
    __shared__ float smemf[SW];

    u16*   rowsU   = (u16*)smemf;                   // [16][C]
    u16*   colsU   = (u16*)(smemf + ROWF);          // [64][CP2]
    float* colNorm = smemf + ROWF + COLF;           // [64]
    float* pdTile  = colNorm + 64;                  // [16][66]
    float* mVal    = smemf;                         // phase 2: [16][8][KNN]
    int*   mIdx    = (int*)smemf + 16 * 8 * KNN;

    int tilesPerB = N / 16;
    int b  = blockIdx.x / tilesPerB;
    int n0 = (blockIdx.x % tilesPerB) * 16;
    int t  = threadIdx.x;
    int wave = t >> 5, lane = t & 31;
    int hl = lane >> 4, lm = lane & 15;
    int r = t >> 3, s = t & 7;                      // insertion mapping
    const u16* F = fb + (size_t)b * N * C;

    // load 16 center rows (contiguous in (N,C) layout)
    {
        const u16* Frow = F + (size_t)n0 * C;
        for (int e = t; e < 16 * C; e += 128) rowsU[e] = Frow[e];
    }

    float vals[KNN]; int ids[KNN];
#pragma unroll
    for (int q = 0; q < KNN; ++q) { vals[q] = -3.4e38f; ids[q] = 0; }

    for (int ct = 0; ct < N; ct += 64) {
        __syncthreads();
        // load 64-column tile of features
        for (int e = t; e < 64 * C; e += 128) {
            int j = e / C, c = e % C;
            colsU[j * CP2 + c] = F[(size_t)(ct + j) * C + c];
        }
        __syncthreads();
        if (t < 64) {
            float s2 = 0.f;
            for (int c = 0; c < C; ++c) { float v = bf2f(colsU[t * CP2 + c]); s2 += v * v; }
            colNorm[t] = s2;
        }
        // wave w computes cols [w*16, w*16+16) of the 16x64 inner-product tile
        {
            v8f acc = {};
#pragma unroll
            for (int kt = 0; kt < KT; ++kt) {
                u32 au[8], bu[8];
#pragma unroll
                for (int v = 0; v < 8; ++v) {
                    int kA = kt * 32 + ((v < 4) ? (2 * v + 8 * hl) : (16 + 2 * (v - 4) + 8 * hl));
                    u32 lo = (kA     < C) ? rowsU[lm * C + kA]     : 0u;
                    u32 hi = (kA + 1 < C) ? rowsU[lm * C + kA + 1] : 0u;
                    au[v] = lo | (hi << 16);
                    int jj = wave * 16 + lm;
                    int kB = kt * 32 + ((v < 4) ? (2 * v + 8 * hl) : (16 + 2 * (v - 4) + 8 * hl));
                    lo = (kB     < C) ? colsU[jj * CP2 + kB]     : 0u;
                    hi = (kB + 1 < C) ? colsU[jj * CP2 + kB + 1] : 0u;
                    bu[v] = lo | (hi << 16);
                }
                acc = wmma_bf16(au, bu, acc);
            }
#pragma unroll
            for (int v = 0; v < 8; ++v)
                pdTile[(v + 8 * hl) * 66 + wave * 16 + lm] = 2.f * acc[v];
        }
        __syncthreads();
        // guarded top-20 insertion: thread (r,s) owns cols s*8..s*8+7
#pragma unroll
        for (int jj = 0; jj < 8; ++jj) {
            int j = s * 8 + jj;
            float cv = pdTile[r * 66 + j] - colNorm[j];
            if (cv > vals[KNN - 1]) {
                int ci = ct + j;
#pragma unroll
                for (int q = 0; q < KNN; ++q) {
                    if (cv > vals[q]) {
                        float tv = vals[q]; vals[q] = cv; cv = tv;
                        int   ti = ids[q];  ids[q]  = ci; ci = ti;
                    }
                }
            }
        }
    }
    __syncthreads();
#pragma unroll
    for (int q = 0; q < KNN; ++q) {
        mVal[(r * 8 + s) * KNN + q] = vals[q];
        mIdx[(r * 8 + s) * KNN + q] = ids[q];
    }
    __syncthreads();
    if (s == 0) {
        int hp[8];
        for (int u = 0; u < 8; ++u) hp[u] = 0;
        int* orow = idxout + ((size_t)b * N + n0 + r) * KNN;
        for (int q = 0; q < KNN; ++q) {
            float best = -3.5e38f; int bu = 0;
            for (int u = 0; u < 8; ++u) {
                if (hp[u] < KNN) {
                    float v = mVal[(r * 8 + u) * KNN + hp[u]];
                    if (v > best) { best = v; bu = u; }
                }
            }
            orow[q] = mIdx[(r * 8 + bu) * KNN + hp[bu]];
            hp[bu]++;
        }
    }
}

// ---------- edge feature pair: [feat-center (C) | center (C)], elems k,k+1 ----
template<int C>
__device__ __forceinline__ u32 edge_pair(const u16* __restrict__ F, int baseN, int baseNb, int k) {
    u32 lo, hi;
    if (k + 1 < C) {                      // both in diff half: needs float math
        lo = f2bf(bf2f(F[baseNb + k])     - bf2f(F[baseN + k]));
        hi = f2bf(bf2f(F[baseNb + k + 1]) - bf2f(F[baseN + k + 1]));
    } else if (k < C) {                   // straddles the boundary
        lo = f2bf(bf2f(F[baseNb + k]) - bf2f(F[baseN + k]));
        hi = (k + 1 < 2 * C) ? (u32)F[baseN + k + 1 - C] : 0u;
    } else {                              // center half: already bf16, raw copy
        lo = (k     < 2 * C) ? (u32)F[baseN + k - C]     : 0u;
        hi = (k + 1 < 2 * C) ? (u32)F[baseN + k + 1 - C] : 0u;
    }
    return lo | (hi << 16);
}

// ---------- edge conv: WMMA GEMM + BN + LReLU + max over k ----------
// One wave per point. A = 32x(2C) edge rows (rows >=20 duplicate row 0, max-safe).
// B = (2C)xO weights from row-major (O,2C) bf16. Epilogue: lane owns one output
// column; 8-row max in regs + shfl_xor(16) half-merge.
template<int C, int O>
__global__ __launch_bounds__(256) void edgeconv_kernel(const u16* __restrict__ fb,
                                                       const int* __restrict__ idx,
                                                       const u16* __restrict__ Wb,
                                                       const float* __restrict__ g,
                                                       const float* __restrict__ bia,
                                                       u16* __restrict__ outb,
                                                       int B, int N) {
    constexpr int KT = (2 * C + 31) / 32;
    int wave = threadIdx.x >> 5;
    int lane = threadIdx.x & 31;
    int p = blockIdx.x * 8 + wave;
    int b = p / N, n = p % N;
    int hl = lane >> 4, lm = lane & 15;

    const u16* F = fb + (size_t)b * N * C;
    int baseN = n * C;
    const int* irow = idx + ((size_t)b * N + n) * KNN;
    int j0 = lm;
    int j1 = 16 + lm; if (j1 >= KNN) j1 = 0;
    int base0 = irow[j0] * C;
    int base1 = irow[j1] * C;

    // A fragments (ISA 16-bit A 16x32 layout): lane M = lane%16,
    // K = kt*32 + (v<4 ? 2v+8*half : 16+2(v-4)+8*half)
    u32 a0[KT][8], a1[KT][8];
#pragma unroll
    for (int kt = 0; kt < KT; ++kt) {
#pragma unroll
        for (int v = 0; v < 8; ++v) {
            int k = kt * 32 + ((v < 4) ? (2 * v + 8 * hl) : (16 + 2 * (v - 4) + 8 * hl));
            a0[kt][v] = edge_pair<C>(F, baseN, base0, k);
            a1[kt][v] = edge_pair<C>(F, baseN, base1, k);
        }
    }

    float inv = rsqrtf(1.f + 1e-5f);
    for (int nt = 0; nt < O / 16; ++nt) {
        int o = nt * 16 + lm;
        v8f c0 = {}, c1 = {};
#pragma unroll
        for (int kt = 0; kt < KT; ++kt) {
            // B fragment (32x16 layout): K = kt*32 + (v<4 ? 2v+8*half : 16+2(v-4)+8*half)
            u32 bu[8];
#pragma unroll
            for (int v = 0; v < 8; ++v) {
                int k = kt * 32 + ((v < 4) ? (2 * v + 8 * hl) : (16 + 2 * (v - 4) + 8 * hl));
                u32 lo = (k     < 2 * C) ? Wb[o * (2 * C) + k]     : 0u;
                u32 hi = (k + 1 < 2 * C) ? Wb[o * (2 * C) + k + 1] : 0u;
                bu[v] = lo | (hi << 16);
            }
            c0 = wmma_bf16(a0[kt], bu, c0);
            c1 = wmma_bf16(a1[kt], bu, c1);
        }
        float sc = g[o] * inv, bb = bia[o];
        float m = -3.4e38f;
#pragma unroll
        for (int v = 0; v < 8; ++v) {
            float h = c0[v] * sc + bb; h = (h >= 0.f) ? h : SLOPE * h; m = fmaxf(m, h);
            h       = c1[v] * sc + bb; h = (h >= 0.f) ? h : SLOPE * h; m = fmaxf(m, h);
        }
        m = fmaxf(m, __shfl_xor(m, 16, 32));
        if (hl == 0)
            outb[((size_t)b * N + n) * O + o] = f2bf(m);
    }
}

// ---------- final: cat(512) -> 1024 GEMM + BN + LReLU + max over N ----------
__device__ __forceinline__ u32 cat_load(const u16* __restrict__ h1,
                                        const u16* __restrict__ h2,
                                        const u16* __restrict__ h3,
                                        const u16* __restrict__ h4,
                                        int b, int N, int n, int c) {
    size_t pn = (size_t)b * N + n;
    const u16* p; int cc;
    if (c < 64)       { p = h1 + pn * 64;  cc = c; }
    else if (c < 128) { p = h2 + pn * 64;  cc = c - 64; }
    else if (c < 256) { p = h3 + pn * 128; cc = c - 128; }
    else              { p = h4 + pn * 256; cc = c - 256; }
    return *(const u32*)(p + cc);   // cc even, dword aligned
}

__global__ __launch_bounds__(256) void final_kernel(const u16* __restrict__ h1b,
                                                    const u16* __restrict__ h2b,
                                                    const u16* __restrict__ h3b,
                                                    const u16* __restrict__ h4b,
                                                    const u16* __restrict__ W5b,
                                                    const float* __restrict__ g5,
                                                    const float* __restrict__ b5,
                                                    float* __restrict__ out, int B, int N) {
    int b  = blockIdx.x >> 6;     // 64 output tiles of 16
    int ot = blockIdx.x & 63;
    int wave = threadIdx.x >> 5, lane = threadIdx.x & 31;
    int hl = lane >> 4, lm = lane & 15;
    int o = ot * 16 + lm;
    float inv = rsqrtf(1.f + 1e-5f);
    float sc = g5[o] * inv, bb = b5[o];
    float runm = -3.4e38f;

    for (int mt = wave; mt < N / 16; mt += 8) {
        int n = mt * 16 + lm;
        v8f c = {};
#pragma unroll
        for (int kt = 0; kt < 16; ++kt) {           // K = 512
            u32 au[8], bu[8];
#pragma unroll
            for (int v = 0; v < 8; ++v) {
                int kA = kt * 32 + ((v < 4) ? (2 * v + 8 * hl) : (16 + 2 * (v - 4) + 8 * hl));
                au[v] = cat_load(h1b, h2b, h3b, h4b, b, N, n, kA);
                int kB = kt * 32 + ((v < 4) ? (2 * v + 8 * hl) : (16 + 2 * (v - 4) + 8 * hl));
                bu[v] = *(const u32*)(W5b + o * 512 + kB);
            }
            c = wmma_bf16(au, bu, c);
        }
#pragma unroll
        for (int v = 0; v < 8; ++v) {
            float h = c[v] * sc + bb;
            h = (h >= 0.f) ? h : SLOPE * h;
            runm = fmaxf(runm, h);
        }
    }
    runm = fmaxf(runm, __shfl_xor(runm, 16, 32));

    __shared__ float red[8][16];
    if (hl == 0) red[wave][lm] = runm;
    __syncthreads();
    if (threadIdx.x < 16) {
        float m = red[0][threadIdx.x];
#pragma unroll
        for (int w = 1; w < 8; ++w) m = fmaxf(m, red[w][threadIdx.x]);
        out[(size_t)b * 1024 + ot * 16 + threadIdx.x] = m;
    }
}

// ---------- host side ----------
extern "C" void kernel_launch(void* const* d_in, const int* in_sizes, int n_in,
                              void* d_out, int out_size, void* d_ws, size_t ws_size,
                              hipStream_t stream) {
    (void)in_sizes; (void)n_in; (void)out_size; (void)ws_size;
    const int B = 8, N = 2048;

    const float* x  = (const float*)d_in[0];
    const float* W1 = (const float*)d_in[1];
    const float* W2 = (const float*)d_in[2];
    const float* W3 = (const float*)d_in[3];
    const float* W4 = (const float*)d_in[4];
    const float* W5 = (const float*)d_in[5];
    const float* g1 = (const float*)d_in[6];  const float* b1 = (const float*)d_in[7];
    const float* g2 = (const float*)d_in[8];  const float* b2 = (const float*)d_in[9];
    const float* g3 = (const float*)d_in[10]; const float* b3 = (const float*)d_in[11];
    const float* g4 = (const float*)d_in[12]; const float* b4 = (const float*)d_in[13];
    const float* g5 = (const float*)d_in[14]; const float* b5 = (const float*)d_in[15];

    char* ws = (char*)d_ws;
    auto alloc = [&](size_t bytes) -> char* {
        char* p = ws;
        ws += (bytes + 255) & ~(size_t)255;
        return p;
    };
    u16* xb  = (u16*)alloc((size_t)B * N * 3 * 2);
    u16* h1b = (u16*)alloc((size_t)B * N * 64 * 2);
    u16* h2b = (u16*)alloc((size_t)B * N * 64 * 2);
    u16* h3b = (u16*)alloc((size_t)B * N * 128 * 2);
    u16* h4b = (u16*)alloc((size_t)B * N * 256 * 2);
    int* idxb = (int*)alloc((size_t)B * N * KNN * 4);
    u16* Wb1 = (u16*)alloc(64 * 6 * 2);
    u16* Wb2 = (u16*)alloc(64 * 128 * 2);
    u16* Wb3 = (u16*)alloc(128 * 128 * 2);
    u16* Wb4 = (u16*)alloc(256 * 256 * 2);
    u16* Wb5 = (u16*)alloc(1024 * 512 * 2);

    auto cvt = [&](const float* src, u16* dst, int n) {
        f32_to_bf16_kernel<<<(n + 255) / 256, 256, 0, stream>>>(src, dst, n);
    };
    cvt(W1, Wb1, 64 * 6);
    cvt(W2, Wb2, 64 * 128);
    cvt(W3, Wb3, 128 * 128);
    cvt(W4, Wb4, 256 * 256);
    cvt(W5, Wb5, 1024 * 512);

    x_transpose_kernel<<<(B * N + 255) / 256, 256, 0, stream>>>(x, xb, B, N);

    int knnGrid = B * N / 16;
    int ecGrid  = B * N / 8;

    knn_kernel<3><<<knnGrid, 128, 0, stream>>>(xb, idxb, B, N);
    edgeconv_kernel<3, 64><<<ecGrid, 256, 0, stream>>>(xb, idxb, Wb1, g1, b1, h1b, B, N);

    knn_kernel<64><<<knnGrid, 128, 0, stream>>>(h1b, idxb, B, N);
    edgeconv_kernel<64, 64><<<ecGrid, 256, 0, stream>>>(h1b, idxb, Wb2, g2, b2, h2b, B, N);

    knn_kernel<64><<<knnGrid, 128, 0, stream>>>(h2b, idxb, B, N);
    edgeconv_kernel<64, 128><<<ecGrid, 256, 0, stream>>>(h2b, idxb, Wb3, g3, b3, h3b, B, N);

    knn_kernel<128><<<knnGrid, 128, 0, stream>>>(h3b, idxb, B, N);
    edgeconv_kernel<128, 256><<<ecGrid, 256, 0, stream>>>(h3b, idxb, Wb4, g4, b4, h4b, B, N);

    final_kernel<<<B * 64, 256, 0, stream>>>(h1b, h2b, h3b, h4b, Wb5, g5, b5,
                                             (float*)d_out, B, N);
}
